// DonutSwinAttention_28295244546143
// MI455X (gfx1250) — matrix operand order repaired and checked by reading, not compiled, not run on hardware
//
#include <hip/hip_runtime.h>
#include <hip/hip_bf16.h>
#include <stdint.h>

typedef __attribute__((ext_vector_type(16))) __bf16 v16bf;
typedef __attribute__((ext_vector_type(8)))  float  v8f;

#define SEQ  49
#define DIM  384
#define NH   12
#define HD   32
#define SEQP 64          // rows padded to multiple of 16
#define XS   392         // X/Q/K/ctx row stride in halves (mult of 8 -> 16B aligned rows)
#define VTS  72          // Vt row stride in halves (token dim padded)
#define SCS  65          // score row stride in floats (bank-conflict friendly)
#define NHB  3           // heads processed per softmax round

__device__ __forceinline__ unsigned short f2bfu(float f) {
  unsigned u = __builtin_bit_cast(unsigned, f);
  return (unsigned short)((u + 0x7FFFu + ((u >> 16) & 1u)) >> 16);   // RNE
}
__device__ __forceinline__ __bf16 f2bf(float f) {
  unsigned short s = f2bfu(f);
  return __builtin_bit_cast(__bf16, s);
}

// A fragment: 16x32 bf16 tile, row-major source with 'stride' halves per row.
// lane holds row m0+(lane&15); halves 0..7 = K[k0+8*(lane>>4) .. +7],
// halves 8..15 = K[k0+8*(lane>>4)+16 .. +23]   (ISA 7.12.2 16-bit A layout)
__device__ __forceinline__ v16bf load_a(const __bf16* base, int stride, int m0, int k0, int lane) {
  int row = m0 + (lane & 15);
  int kb  = k0 + ((lane >> 4) << 3);
  const __bf16* p = base + row * stride + kb;
  union { v16bf v; uint4 q[2]; } u;
  u.q[0] = *(const uint4*)(p);
  u.q[1] = *(const uint4*)(p + 16);
  return u.v;
}

// B fragment: 32x16 bf16 tile from K-contiguous ("transposed") storage B[n][k].
// lane holds col n0+(lane&15); halves 0..7 = K[k0+8*(lane>>4) .. +7], halves 8..15 = K[k0+8*(lane>>4)+16 .. +23]
__device__ __forceinline__ v16bf load_b(const __bf16* base, int stride, int n0, int k0, int lane) {
  int n  = n0 + (lane & 15);
  int kk = k0 + ((lane >> 4) << 3);
  const __bf16* p = base + n * stride + kk;
  union { v16bf v; uint4 q[2]; } u;
  u.q[0] = *(const uint4*)(p);
  u.q[1] = *(const uint4*)(p + 16);
  return u.v;
}

// ---------------- prep: transpose+cvt weights to bf16; pad rel-pos bias ----------------
__global__ void swin_prep(const float* __restrict__ qW, const float* __restrict__ kW,
                          const float* __restrict__ vW, const float* __restrict__ oW,
                          const float* __restrict__ bt, const int* __restrict__ ri,
                          unsigned short* __restrict__ Wt, float* __restrict__ rpb) {
  int idx = blockIdx.x * blockDim.x + threadIdx.x;
  const int NW = 4 * DIM * DIM;
  if (idx < NW) {
    int p = idx / (DIM * DIM);
    int r = idx % (DIM * DIM);
    int k = r / DIM, n = r % DIM;
    const float* W = (p == 0) ? qW : (p == 1) ? kW : (p == 2) ? vW : oW;
    Wt[p * (DIM * DIM) + n * DIM + k] = f2bfu(W[k * DIM + n]);   // Wt[p][n][k]
  } else {
    int j = idx - NW;
    if (j < NH * SEQP * SEQP) {
      int h = j >> 12, m = (j >> 6) & 63, n = j & 63;
      float v = 0.f;
      if (m < SEQ && n < SEQ) v = bt[ri[m * SEQ + n] * NH + h];
      rpb[j] = v;
    }
  }
}

// ---------------- fused per-window attention ----------------
__global__ __launch_bounds__(384, 1)
void swin_attn_fused(const float* __restrict__ hs, const float* __restrict__ amask,
                     const float* __restrict__ qbias, const float* __restrict__ kbias,
                     const float* __restrict__ vbias, const float* __restrict__ obias,
                     const unsigned short* __restrict__ Wt_u, const float* __restrict__ rpb,
                     float* __restrict__ out) {
  extern __shared__ char smem[];
  __bf16* Xb = (__bf16*)smem;              // [64][392]  X, later reused as ctx
  __bf16* Qb = Xb + SEQP * XS;             // [64][392]
  __bf16* Kb = Qb + SEQP * XS;             // [64][392]
  __bf16* Vt = Kb + SEQP * XS;             // [384][72]  V transposed (col-major)
  float*  sc = (float*)(Vt + DIM * VTS);   // [3][64][65] f32 scores
  __bf16* at = (__bf16*)(sc + NHB * SEQP * SCS); // [3][64][64] bf16 attn

  const __bf16* Wt = (const __bf16*)Wt_u;
  const int tid  = threadIdx.x;
  const int lane = tid & 31;
  const int wave = tid >> 5;
  const int b    = blockIdx.x;

  // ---- stage 1: X -> bf16 LDS (+ zero pad rows) ----
  const float* xg = hs + (size_t)b * (SEQ * DIM);
  for (int c4 = tid; c4 < SEQ * DIM / 4; c4 += 384) {
    int f = c4 * 4;
    int row = f / DIM, col = f % DIM;
    float4 x = *(const float4*)(xg + f);
    uint2 v;
    v.x = (unsigned)f2bfu(x.x) | ((unsigned)f2bfu(x.y) << 16);
    v.y = (unsigned)f2bfu(x.z) | ((unsigned)f2bfu(x.w) << 16);
    *(uint2*)(Xb + row * XS + col) = v;
  }
  for (int i = tid; i < (SEQP - SEQ) * XS; i += 384) {
    int row = SEQ + i / XS, col = i % XS;
    Xb[row * XS + col] = f2bf(0.f);
  }
  __syncthreads();

  // ---- stage 2: QKV projections, 4 M-tiles per job (B loaded once per k-step) ----
  for (int job = wave; job < 72; job += 12) {
    int p  = job / 24;
    int nt = job % 24;
    const __bf16* W = Wt + (size_t)p * (DIM * DIM);
    __builtin_prefetch(W + (nt * 16) * DIM, 0, 1);
    v8f acc[4] = {{}, {}, {}, {}};
    for (int ks = 0; ks < 12; ++ks) {
      v16bf bb = load_b(W, DIM, nt * 16, ks * 32, lane);
#pragma unroll
      for (int mt = 0; mt < 4; ++mt) {
        v16bf a = load_a(Xb, XS, mt * 16, ks * 32, lane);
        acc[mt] = __builtin_amdgcn_wmma_f32_16x16x32_bf16(false, a, false, bb, (short)0,
                                                          acc[mt], false, false);
      }
    }
    int col = nt * 16 + (lane & 15);
    const float* bias = (p == 0) ? qbias : (p == 1) ? kbias : vbias;
    float bv = bias[col];
#pragma unroll
    for (int mt = 0; mt < 4; ++mt) {
      int rb = mt * 16 + ((lane >> 4) << 3);
      if (p == 2) {
        for (int i = 0; i < 8; ++i) Vt[col * VTS + rb + i] = f2bf(acc[mt][i] + bv);
      } else {
        __bf16* dst = (p == 0) ? Qb : Kb;
        for (int i = 0; i < 8; ++i) dst[(rb + i) * XS + col] = f2bf(acc[mt][i] + bv);
      }
    }
  }
  __syncthreads();

  // ---- stage 3: attention, three heads per round ----
  const float* mrow = amask + (size_t)(b & 63) * (SEQ * SEQ);
  for (int hp = 0; hp < NH / NHB; ++hp) {
    // scores: job = (hh, nt); K-tile (B) shared across 4 Q M-tiles; 12 jobs = 1/wave
    {
      int job = wave;                     // 0..11
      int hh = job >> 2;                  // 0..2
      int nt = job & 3;                   // 0..3
      int head = hp * NHB + hh;
      v16bf bb = load_b(Kb, XS, nt * 16, head * HD, lane);
      int n  = nt * 16 + (lane & 15);
      float* sh = sc + hh * (SEQP * SCS);
      const float* rp = rpb + head * (SEQP * SEQP);
#pragma unroll
      for (int mt = 0; mt < 4; ++mt) {
        v16bf a = load_a(Qb, XS, mt * 16, head * HD, lane);
        v8f c = {};
        c = __builtin_amdgcn_wmma_f32_16x16x32_bf16(false, a, false, bb, (short)0, c, false, false);
        int rb = mt * 16 + ((lane >> 4) << 3);
        if (n >= SEQ) {
          for (int i = 0; i < 8; ++i) sh[(rb + i) * SCS + n] = -1e30f;
        } else {
          for (int i = 0; i < 8; ++i) {
            int m = rb + i;
            float s = c[i] * 0.1767766952966369f + rp[m * SEQP + n];
            if (m < SEQ) s += mrow[m * SEQ + n];
            sh[m * SCS + n] = s;
          }
        }
      }
    }
    __syncthreads();
    // softmax: one thread per row (192 rows); padded cols hold -1e30 -> exp 0
    if (tid < NHB * SEQP) {
      int hh = tid >> 6, m = tid & 63;
      float* srow = sc + hh * (SEQP * SCS) + m * SCS;
      float mx = -1e30f;
      for (int n = 0; n < SEQP; ++n) mx = fmaxf(mx, srow[n]);
      float sum = 0.f;
      for (int n = 0; n < SEQP; ++n) { float e = __expf(srow[n] - mx); srow[n] = e; sum += e; }
      float rinv = 1.f / sum;
      __bf16* arow = at + hh * (SEQP * 64) + m * 64;
      for (int n = 0; n < SEQP; ++n) arow[n] = f2bf(srow[n] * rinv);
    }
    __syncthreads();
    // ctx = attn @ V: job = (hh, mt); A (attn) shared across the 2 V N-tiles; 12 jobs = 1/wave
    {
      int job = wave;
      int hh = job >> 2;                  // 0..2
      int mt = job & 3;                   // 0..3
      int head = hp * NHB + hh;
      v8f acc[2] = {{}, {}};
      for (int ks = 0; ks < 2; ++ks) {
        v16bf a = load_a(at + hh * (SEQP * 64), 64, mt * 16, ks * 32, lane);
#pragma unroll
        for (int nt2 = 0; nt2 < 2; ++nt2) {
          v16bf bb = load_b(Vt, VTS, head * HD + nt2 * 16, ks * 32, lane);
          acc[nt2] = __builtin_amdgcn_wmma_f32_16x16x32_bf16(false, a, false, bb, (short)0,
                                                             acc[nt2], false, false);
        }
      }
      int rb = mt * 16 + ((lane >> 4) << 3);
#pragma unroll
      for (int nt2 = 0; nt2 < 2; ++nt2) {
        int col = head * HD + nt2 * 16 + (lane & 15);
        for (int i = 0; i < 8; ++i) Xb[(rb + i) * XS + col] = f2bf(acc[nt2][i]);
      }
    }
    __syncthreads();
  }

  // ---- stage 4: output projection, 4 M-tiles per job ----
  const __bf16* oW = Wt + (size_t)3 * (DIM * DIM);
  float* og = out + (size_t)b * (SEQ * DIM);
  for (int job = wave; job < 24; job += 12) {
    int nt = job;
    __builtin_prefetch(oW + (nt * 16) * DIM, 0, 1);
    v8f acc[4] = {{}, {}, {}, {}};
    for (int ks = 0; ks < 12; ++ks) {
      v16bf bb = load_b(oW, DIM, nt * 16, ks * 32, lane);
#pragma unroll
      for (int mt = 0; mt < 4; ++mt) {
        v16bf a = load_a(Xb, XS, mt * 16, ks * 32, lane);
        acc[mt] = __builtin_amdgcn_wmma_f32_16x16x32_bf16(false, a, false, bb, (short)0,
                                                          acc[mt], false, false);
      }
    }
    int col = nt * 16 + (lane & 15);
    float bv = obias[col];
#pragma unroll
    for (int mt = 0; mt < 4; ++mt) {
      int rb = mt * 16 + ((lane >> 4) << 3);
      for (int i = 0; i < 8; ++i) {
        int m = rb + i;
        if (m < SEQ) og[m * DIM + col] = acc[mt][i] + bv;
      }
    }
  }
}

extern "C" void kernel_launch(void* const* d_in, const int* in_sizes, int n_in,
                              void* d_out, int out_size, void* d_ws, size_t ws_size,
                              hipStream_t stream) {
  const float* hs = (const float*)d_in[0];
  const float* am = (const float*)d_in[1];
  const float* qW = (const float*)d_in[2];
  const float* qb = (const float*)d_in[3];
  const float* kW = (const float*)d_in[4];
  const float* kb = (const float*)d_in[5];
  const float* vW = (const float*)d_in[6];
  const float* vb = (const float*)d_in[7];
  const float* oW = (const float*)d_in[8];
  const float* ob = (const float*)d_in[9];
  const float* bt = (const float*)d_in[10];
  const int*   ri = (const int*)d_in[11];

  unsigned short* Wt = (unsigned short*)d_ws;                     // 4*384*384 bf16
  float* rpb = (float*)((char*)d_ws + (size_t)4 * DIM * DIM * 2); // 12*64*64 f32
  float* out = (float*)d_out;

  int total = 4 * DIM * DIM + NH * SEQP * SEQP;
  swin_prep<<<(total + 255) / 256, 256, 0, stream>>>(qW, kW, vW, oW, bt, ri, Wt, rpb);

  size_t lds = (size_t)(3 * SEQP * XS + DIM * VTS) * 2    // X/Q/K + Vt (bf16)
             + (size_t)(NHB * SEQP * SCS) * 4             // scores (f32)
             + (size_t)(NHB * SEQP * 64) * 2;             // attn (bf16)
  swin_attn_fused<<<2048, 384, lds, stream>>>(hs, am, qb, kb, vb, ob, Wt, rpb, out);
}
